// LiquidNeuron_18769007084243
// MI455X (gfx1250) — matrix operand order repaired and checked by reading, not compiled, not run on hardware
//
#include <hip/hip_runtime.h>
#include <hip/hip_bf16.h>

// ---------------------------------------------------------------------------
// LiquidNeuron fused pipeline for gfx1250 (MI455X), wave32 + WMMA bf16
// + async global->LDS staged GEMM (double-buffered, ASYNCcnt-tracked).
// ---------------------------------------------------------------------------

typedef __bf16        bf16x16 __attribute__((ext_vector_type(16)));
typedef float         f32x8   __attribute__((ext_vector_type(8)));
typedef unsigned int  u32x4   __attribute__((ext_vector_type(4)));

#define BATCH   8192
#define IN_DIM  4096
#define OUT_DIM 4096

// LDS tile row stride: 64B of K-data + 16B pad -> 16B aligned, conflict-free
#define LDS_ROW_U16 40           // 80 bytes
#define TILE_U16    (128 * LDS_ROW_U16)

union FragAB {
    bf16x16 v;
    u32x4   u[2];
};

__device__ __forceinline__ unsigned short f32_to_bf16(float f) {
    unsigned int u = __float_as_uint(f);
    unsigned int r = (u + 0x7FFFu + ((u >> 16) & 1u)) >> 16;
    return (unsigned short)r;
}

// CDNA5 async global -> LDS copy, 16B per lane (tracked by ASYNCcnt).
__device__ __forceinline__ void async_copy_b128(unsigned lds_off, const void* gptr) {
    asm volatile("global_load_async_to_lds_b128 %0, %1, off"
                 :: "v"(lds_off), "v"(gptr) : "memory");
}

// ---------------------------------------------------------------------------
// 1) x (f32) -> x_bf16, plus global sum of x^2 (for x_norm)
// ---------------------------------------------------------------------------
__global__ void k_convert_x(const float* __restrict__ x,
                            unsigned short* __restrict__ xbf,
                            float* __restrict__ sumsq_accum) {
    __shared__ float red[256];
    const int tid  = threadIdx.x;
    const size_t base = ((size_t)blockIdx.x * 256 + tid) * 8;
    float4 a = *(const float4*)(x + base);
    float4 b = *(const float4*)(x + base + 4);
    float s2 = a.x*a.x + a.y*a.y + a.z*a.z + a.w*a.w
             + b.x*b.x + b.y*b.y + b.z*b.z + b.w*b.w;

    u32x4 packed;
    packed.x = (unsigned)f32_to_bf16(a.x) | ((unsigned)f32_to_bf16(a.y) << 16);
    packed.y = (unsigned)f32_to_bf16(a.z) | ((unsigned)f32_to_bf16(a.w) << 16);
    packed.z = (unsigned)f32_to_bf16(b.x) | ((unsigned)f32_to_bf16(b.y) << 16);
    packed.w = (unsigned)f32_to_bf16(b.z) | ((unsigned)f32_to_bf16(b.w) << 16);
    *(u32x4*)(xbf + base) = packed;

    red[tid] = s2;
    __syncthreads();
    for (int off = 128; off; off >>= 1) {
        if (tid < off) red[tid] += red[tid + off];
        __syncthreads();
    }
    if (tid == 0) atomicAdd(sumsq_accum, red[0]);
}

// ---------------------------------------------------------------------------
// 2) W_sum = bf16(W_slow + W_fast)
// ---------------------------------------------------------------------------
__global__ void k_wsum(const float* __restrict__ ws,
                       const float* __restrict__ wf,
                       unsigned short* __restrict__ wsum) {
    const size_t base = ((size_t)blockIdx.x * 256 + threadIdx.x) * 8;
    float4 a0 = *(const float4*)(ws + base);
    float4 a1 = *(const float4*)(ws + base + 4);
    float4 b0 = *(const float4*)(wf + base);
    float4 b1 = *(const float4*)(wf + base + 4);
    float v[8] = { a0.x+b0.x, a0.y+b0.y, a0.z+b0.z, a0.w+b0.w,
                   a1.x+b1.x, a1.y+b1.y, a1.z+b1.z, a1.w+b1.w };
    u32x4 packed;
    packed.x = (unsigned)f32_to_bf16(v[0]) | ((unsigned)f32_to_bf16(v[1]) << 16);
    packed.y = (unsigned)f32_to_bf16(v[2]) | ((unsigned)f32_to_bf16(v[3]) << 16);
    packed.z = (unsigned)f32_to_bf16(v[4]) | ((unsigned)f32_to_bf16(v[5]) << 16);
    packed.w = (unsigned)f32_to_bf16(v[6]) | ((unsigned)f32_to_bf16(v[7]) << 16);
    *(u32x4*)(wsum + base) = packed;
}

// ---------------------------------------------------------------------------
// 3) NT GEMM, LDS-staged:  C[m,n] = alpha * sum_k A[m*K+k] * Bt[n*K+k]
//    block = 256 threads = 8 waves (2 x 4). block tile 128x128,
//    wave tile 64x32 (4 A-frags x 2 B-frags = 8 wmma per K=32 step).
//    A/B tiles double-buffered in LDS, filled with async b128 copies.
//    Hot loop is straight-line: issue next tile -> s_wait_asynccnt 4 ->
//    barrier -> ds_load frags + 8 wmma -> barrier. Last stage peeled.
// ---------------------------------------------------------------------------
__device__ __forceinline__ void store_tile(const f32x8& acc, float* __restrict__ C,
                                           int m, int n, int ldc, int half,
                                           int l16, float alpha) {
#pragma unroll
    for (int r = 0; r < 8; ++r) {
        C[(size_t)(m + r + half * 8) * ldc + n + l16] = alpha * acc[r];
    }
}

__global__ void __launch_bounds__(256)
k_gemm_bf16_nt(const unsigned short* __restrict__ A,
               const unsigned short* __restrict__ Bt,
               float* __restrict__ C,
               int K, int N, float alpha) {
    __shared__ __align__(16) unsigned short smA[2][TILE_U16];
    __shared__ __align__(16) unsigned short smB[2][TILE_U16];

    const int tid  = threadIdx.x;
    const int lane = tid & 31;
    const int wave = tid >> 5;
    const int wm   = wave >> 2;      // 0..1  (64-row half)
    const int wn   = wave & 3;       // 0..3  (32-col quarter)
    const int half = lane >> 4;
    const int l16  = lane & 15;

    const int mBlock = blockIdx.y * 128;
    const int nBlock = blockIdx.x * 128;

    // ---- cooperative copy assignment: each thread moves 32B of one row ----
    const int rowc = tid >> 1;               // 0..127
    const int chk  = (tid & 1) * 2;          // chunk pair {0,1} or {2,3}
    const unsigned short* gA = A  + (size_t)(mBlock + rowc) * K + chk * 8;
    const unsigned short* gB = Bt + (size_t)(nBlock + rowc) * K + chk * 8;
    const unsigned ldsA = (unsigned)(uintptr_t)(&smA[0][0]) + rowc * 80 + chk * 16;
    const unsigned ldsB = (unsigned)(uintptr_t)(&smB[0][0]) + rowc * 80 + chk * 16;
    const unsigned bufStride = TILE_U16 * 2; // bytes per buffer

    f32x8 acc[4][2];
#pragma unroll
    for (int i = 0; i < 4; ++i)
#pragma unroll
        for (int j = 0; j < 2; ++j) acc[i][j] = {};

    auto compute_stage = [&](int buf) {
        FragAB a[4], b[2];
#pragma unroll
        for (int fi = 0; fi < 4; ++fi) {
            const unsigned short* p =
                &smA[buf][(wm * 64 + fi * 16 + l16) * LDS_ROW_U16 + half * 8];
            a[fi].u[0] = *(const u32x4*)(p);
            a[fi].u[1] = *(const u32x4*)(p + 16);
        }
#pragma unroll
        for (int fj = 0; fj < 2; ++fj) {
            const unsigned short* p =
                &smB[buf][(wn * 32 + fj * 16 + l16) * LDS_ROW_U16 + half * 8];
            b[fj].u[0] = *(const u32x4*)(p);
            b[fj].u[1] = *(const u32x4*)(p + 16);
        }
#pragma unroll
        for (int fi = 0; fi < 4; ++fi)
#pragma unroll
            for (int fj = 0; fj < 2; ++fj)
                acc[fi][fj] = __builtin_amdgcn_wmma_f32_16x16x32_bf16(
                    false, a[fi].v, false, b[fj].v, (short)0, acc[fi][fj],
                    false, false);
    };

    const int nstage = K >> 5;

    // prologue: stage 0 into buffer 0
    async_copy_b128(ldsA, gA);
    async_copy_b128(ldsA + 16, gA + 8);
    async_copy_b128(ldsB, gB);
    async_copy_b128(ldsB + 16, gB + 8);

    const unsigned short* gAn = gA + 32;   // stage s+1 source (advance 64B/stage)
    const unsigned short* gBn = gB + 32;

    // steady state: last stage peeled so the hot loop is branch-free
    for (int s = 0; s < nstage - 1; ++s) {
        const int buf = s & 1;
        const unsigned bo = (buf ^ 1) * bufStride;
        async_copy_b128(ldsA + bo, gAn);
        async_copy_b128(ldsA + bo + 16, gAn + 8);
        async_copy_b128(ldsB + bo, gBn);
        async_copy_b128(ldsB + bo + 16, gBn + 8);
        gAn += 32;
        gBn += 32;
        asm volatile("s_wait_asynccnt 0x4" ::: "memory");
        __syncthreads();          // stage-s tiles visible to all waves
        compute_stage(buf);
        __syncthreads();          // readers done before next stage overwrites
    }

    // epilogue: final stage
    asm volatile("s_wait_asynccnt 0x0" ::: "memory");
    __syncthreads();
    compute_stage((nstage - 1) & 1);

    const int mW = mBlock + wm * 64;
    const int nW = nBlock + wn * 32;
#pragma unroll
    for (int fi = 0; fi < 4; ++fi)
#pragma unroll
        for (int fj = 0; fj < 2; ++fj)
            store_tile(acc[fi][fj], C, mW + fi * 16, nW + fj * 16, N,
                       half, l16, alpha);
}

// ---------------------------------------------------------------------------
// 4) per-column partial sums of pre_act (mean / std over batch)
// ---------------------------------------------------------------------------
__global__ void k_colstats(const float* __restrict__ pre,
                           float* __restrict__ colsum,
                           float* __restrict__ colsumsq) {
    const int j  = blockIdx.x * 256 + threadIdx.x;  // column
    const int r0 = blockIdx.y * 256;                // row chunk
    float s = 0.f, s2 = 0.f;
    for (int r = r0; r < r0 + 256; ++r) {
        float v = pre[(size_t)r * OUT_DIM + j];
        s += v;
        s2 += v * v;
    }
    atomicAdd(&colsum[j], s);
    atomicAdd(&colsumsq[j], s2);
}

// ---------------------------------------------------------------------------
// 5) per-column plasticity controller MLP (2-16-1) -> eff[], sum(eff)
// ---------------------------------------------------------------------------
__global__ void k_mlp(const float* __restrict__ colsum,
                      const float* __restrict__ colsumsq,
                      const float* __restrict__ w1, const float* __restrict__ b1,
                      const float* __restrict__ w2, const float* __restrict__ b2,
                      const float* __restrict__ gp,
                      float* __restrict__ eff, float* __restrict__ eff_sum) {
    const int j = blockIdx.x * 256 + threadIdx.x;
    const float invB = 1.0f / (float)BATCH;
    float mean = colsum[j] * invB;
    float var  = (colsumsq[j] - (float)BATCH * mean * mean) / ((float)BATCH - 1.0f);
    float sd   = sqrtf(fmaxf(var, 0.0f)) + 1e-6f;

    float acc = b2[0];
#pragma unroll
    for (int i = 0; i < 16; ++i) {
        float h = tanhf(w1[i * 2 + 0] * mean + w1[i * 2 + 1] * sd + b1[i]);
        acc += w2[i] * h;
    }
    float lp = 1.0f / (1.0f + expf(-acc));
    float e  = gp[0] * lp;      // PRED_ERR == 0
    eff[j] = e;
    atomicAdd(eff_sum, e);
}

// ---------------------------------------------------------------------------
// 6) row LayerNorm (eps 1e-5) + 5*tanh(./5) -> out (f32, d_out)
// ---------------------------------------------------------------------------
__global__ void __launch_bounds__(256)
k_layernorm(const float* __restrict__ pre,
            const float* __restrict__ gamma,
            const float* __restrict__ beta,
            float* __restrict__ out) {
    __shared__ float rs[256], rs2[256];
    const int row = blockIdx.x;
    const int tid = threadIdx.x;
    const float4* p = (const float4*)(pre + (size_t)row * OUT_DIM);

    float4 v[4];
    float s = 0.f, s2 = 0.f;
#pragma unroll
    for (int i = 0; i < 4; ++i) {
        v[i] = p[tid + i * 256];
        s  += v[i].x + v[i].y + v[i].z + v[i].w;
        s2 += v[i].x*v[i].x + v[i].y*v[i].y + v[i].z*v[i].z + v[i].w*v[i].w;
    }
    rs[tid] = s; rs2[tid] = s2;
    __syncthreads();
    for (int off = 128; off; off >>= 1) {
        if (tid < off) { rs[tid] += rs[tid + off]; rs2[tid] += rs2[tid + off]; }
        __syncthreads();
    }
    const float inv = 1.0f / (float)OUT_DIM;
    const float mu  = rs[0] * inv;
    const float var = rs2[0] * inv - mu * mu;
    const float rstd = rsqrtf(var + 1e-5f);

    float4* o = (float4*)(out + (size_t)row * OUT_DIM);
#pragma unroll
    for (int i = 0; i < 4; ++i) {
        int col = (tid + i * 256) * 4;
        float4 g = *(const float4*)(gamma + col);
        float4 bt = *(const float4*)(beta + col);
        float4 r;
        r.x = 5.0f * tanhf(((v[i].x - mu) * rstd * g.x + bt.x) * 0.2f);
        r.y = 5.0f * tanhf(((v[i].y - mu) * rstd * g.y + bt.y) * 0.2f);
        r.z = 5.0f * tanhf(((v[i].z - mu) * rstd * g.z + bt.z) * 0.2f);
        r.w = 5.0f * tanhf(((v[i].w - mu) * rstd * g.w + bt.w) * 0.2f);
        o[tid + i * 256] = r;
    }
}

// ---------------------------------------------------------------------------
// 7) f32 [R,C] -> bf16 transpose [C,R] via LDS 32x32 tile
// ---------------------------------------------------------------------------
__global__ void k_transpose_f32_bf16(const float* __restrict__ src,
                                     unsigned short* __restrict__ dst,
                                     int R, int C) {
    __shared__ float tile[32][33];
    const int tx = threadIdx.x, ty = threadIdx.y;
    const int x = blockIdx.x * 32 + tx;
#pragma unroll
    for (int i = 0; i < 4; ++i) {
        int y = blockIdx.y * 32 + ty + i * 8;
        tile[ty + i * 8][tx] = src[(size_t)y * C + x];
    }
    __syncthreads();
    const int xo = blockIdx.y * 32 + tx;
#pragma unroll
    for (int i = 0; i < 4; ++i) {
        int yo = blockIdx.x * 32 + ty + i * 8;
        dst[(size_t)yo * R + xo] = f32_to_bf16(tile[tx][ty + i * 8]);
    }
}

// ---------------------------------------------------------------------------
// 8) Hebbian fast-weight update -> new_W_fast (second region of d_out)
// ---------------------------------------------------------------------------
__global__ void k_update(const float* __restrict__ corr,
                         const float* __restrict__ wf,
                         const float* __restrict__ eff,
                         const float* __restrict__ stats,  // [0]=sum(x^2), [1]=sum(eff)
                         float* __restrict__ out_wf) {
    const size_t idx = (size_t)blockIdx.x * 256 + threadIdx.x;
    const float xn   = stats[0] * (1.0f / (float)BATCH) + 1e-6f;
    const bool train = (stats[1] * (1.0f / (float)OUT_DIM)) > 0.001f;
    const int o = (int)(idx >> 12);  // / IN_DIM (4096)

    float w = wf[idx];
    float delta = corr[idx] / xn - 0.2f * w;
    delta = fminf(fmaxf(delta, -0.05f), 0.05f);
    float lr = eff[o] * 0.015f;
    float upd = (w + delta * lr) * 0.999f;
    out_wf[idx] = train ? upd : w;
}

// ---------------------------------------------------------------------------
// launch
// ---------------------------------------------------------------------------
extern "C" void kernel_launch(void* const* d_in, const int* in_sizes, int n_in,
                              void* d_out, int out_size, void* d_ws, size_t ws_size,
                              hipStream_t stream) {
    const float* x      = (const float*)d_in[0];
    const float* W_slow = (const float*)d_in[1];
    const float* W_fast = (const float*)d_in[2];
    const float* gamma  = (const float*)d_in[3];
    const float* beta   = (const float*)d_in[4];
    const float* w1     = (const float*)d_in[5];
    const float* b1     = (const float*)d_in[6];
    const float* w2     = (const float*)d_in[7];
    const float* b2     = (const float*)d_in[8];
    const float* gp     = (const float*)d_in[9];

    float* out_act = (float*)d_out;                                   // [B, O]
    float* out_wf  = (float*)d_out + (size_t)BATCH * OUT_DIM;         // [O, I]

    char* ws = (char*)d_ws;
    const size_t OFF_PREACT = 0;                                      // B*O*4 (reused for corr)
    const size_t OFF_XBF    = OFF_PREACT + (size_t)BATCH * OUT_DIM * 4;
    const size_t OFF_XT     = OFF_XBF + (size_t)BATCH * IN_DIM * 2;
    const size_t OFF_WSUM   = OFF_XT + (size_t)IN_DIM * BATCH * 2;
    const size_t OFF_OUTT   = OFF_WSUM + (size_t)OUT_DIM * IN_DIM * 2;
    const size_t OFF_STATS  = OFF_OUTT + (size_t)OUT_DIM * BATCH * 2;

    float* preact = (float*)(ws + OFF_PREACT);
    float* corr   = (float*)(ws + OFF_PREACT);   // aliased: pre_act dead after LN
    unsigned short* xbf  = (unsigned short*)(ws + OFF_XBF);
    unsigned short* xT   = (unsigned short*)(ws + OFF_XT);
    unsigned short* wsum = (unsigned short*)(ws + OFF_WSUM);
    unsigned short* outT = (unsigned short*)(ws + OFF_OUTT);
    float* stats    = (float*)(ws + OFF_STATS);  // [0]=sum(x^2) [1]=sum(eff)
    float* colsum   = stats + 2;
    float* colsumsq = colsum + OUT_DIM;
    float* eff      = colsumsq + OUT_DIM;

    // zero accumulators (sumsq, effsum, colsum, colsumsq)
    hipMemsetAsync(stats, 0, (2 + 2 * OUT_DIM) * sizeof(float), stream);

    // 1) x -> bf16 + sum(x^2)
    k_convert_x<<<(size_t)BATCH * IN_DIM / (256 * 8), 256, 0, stream>>>(x, xbf, &stats[0]);

    // 2) W_slow + W_fast -> bf16
    k_wsum<<<(size_t)OUT_DIM * IN_DIM / (256 * 8), 256, 0, stream>>>(W_slow, W_fast, wsum);

    // 3) pre_act = x @ W_sum^T   (M=8192, N=4096, K=4096)
    k_gemm_bf16_nt<<<dim3(OUT_DIM / 128, BATCH / 128), 256, 0, stream>>>(
        xbf, wsum, preact, IN_DIM, OUT_DIM, 1.0f);

    // 4) column stats
    k_colstats<<<dim3(OUT_DIM / 256, BATCH / 256), 256, 0, stream>>>(preact, colsum, colsumsq);

    // 5) plasticity MLP -> eff[], sum(eff)
    k_mlp<<<OUT_DIM / 256, 256, 0, stream>>>(colsum, colsumsq, w1, b1, w2, b2, gp,
                                             eff, &stats[1]);

    // 6) LayerNorm + soft clip -> out (f32)
    k_layernorm<<<BATCH, 256, 0, stream>>>(preact, gamma, beta, out_act);

    // 7) transposes for GEMM2: xT [I,B] bf16, outT [O,B] bf16
    k_transpose_f32_bf16<<<dim3(IN_DIM / 32, BATCH / 32), dim3(32, 8), 0, stream>>>(
        x, xT, BATCH, IN_DIM);
    k_transpose_f32_bf16<<<dim3(OUT_DIM / 32, BATCH / 32), dim3(32, 8), 0, stream>>>(
        out_act, outT, BATCH, OUT_DIM);

    // 8) correlation = out^T @ x / B   (M=4096, N=4096, K=8192)
    k_gemm_bf16_nt<<<dim3(IN_DIM / 128, OUT_DIM / 128), 256, 0, stream>>>(
        outT, xT, corr, BATCH, IN_DIM, 1.0f / (float)BATCH);

    // 9) fast-weight update
    k_update<<<(size_t)OUT_DIM * IN_DIM / 256, 256, 0, stream>>>(
        corr, W_fast, eff, stats, out_wf);
}
